// UNetV2_74096775790901
// MI455X (gfx1250) — compile-verified, matrix operand order checked by reading, not run on hardware
//
#include <hip/hip_runtime.h>

typedef __attribute__((ext_vector_type(16))) _Float16 v16h;
typedef __attribute__((ext_vector_type(8)))  _Float16 v8h;
typedef __attribute__((ext_vector_type(8)))  float    v8f;
typedef unsigned int u32x4 __attribute__((ext_vector_type(4)));
typedef int          i32x4 __attribute__((ext_vector_type(4)));
typedef int          i32x8 __attribute__((ext_vector_type(8)));

#if __has_builtin(__builtin_amdgcn_tensor_load_to_lds) && __has_builtin(__builtin_amdgcn_s_wait_tensorcnt)
#define HAVE_TDM 1
#else
#define HAVE_TDM 0
#endif

struct ConvArgs {
    const float*    featA;   // first concat half (or sole input)
    const float*    featB;   // second concat half (nullptr if none)
    const int*      nbr;     // Nout x 27 neighbor indices (-1 = missing)
    const _Float16* W;       // PRE-SWIZZLED fragment-major f16: 27 x (KP*COUT)
    const float*    scale;   // Cout
    const float*    bias;    // Cout
    const float*    res;     // residual (mode 1), Nout x Cout
    float*          out;     // Nout x Cout
    int Nout;
    int mode;                // 0=relu, 1=res+relu, 2=relu+reduce(cat)
};

// ---- repack f32 weights -> f16, fragment-major, K padded to KP ----
template <int CTOT, int COUT>
__global__ void repack_kernel(const float* __restrict__ src,
                              _Float16* __restrict__ dst, int nSlab) {
    constexpr int KP  = (CTOT + 31) & ~31;
    constexpr int NNT = COUT / 16;
    constexpr int WK  = KP * COUT;
    int j = blockIdx.x * 256 + threadIdx.x;
    if (j >= nSlab * WK) return;
    int slab = j / WK;
    int jj   = j - slab * WK;
    int q  = jj & 15;
    int ln = (jj >> 4) & 31;
    int r  = jj >> 9;
    int nt = r % NNT;
    int ks = r / NNT;
    int K  = ((q >> 3) << 4) + ((ln >> 4) << 3) + (q & 7);
    int ci = ks * 32 + K;
    int co = nt * 16 + (ln & 15);
    dst[j] = (ci < CTOT)
                 ? (_Float16)src[(size_t)slab * CTOT * COUT + ci * COUT + co]
                 : (_Float16)0.f;
}

// ---- Tensor Data Mover descriptor (D#) helpers ----
__device__ inline void dset(unsigned* w, int lo, int len, unsigned long long v) {
    for (int i = 0; i < len; ++i)
        if ((v >> i) & 1ull) w[(lo + i) >> 5] |= 1u << ((lo + i) & 31);
}

// Async DMA: copy `rows` x `elems` f16 (contiguous, stride = elems) from global
// to LDS via TENSOR_LOAD_TO_LDS. Completion via s_wait_tensorcnt.
__device__ inline void tdm_load_f16(const _Float16* gsrc, _Float16* ldsDst,
                                    unsigned elems, unsigned rows) {
#if HAVE_TDM
    unsigned g0w[4] = {}, g1w[8] = {}, g2w[4] = {}, g3w[4] = {};
    dset(g0w, 0, 2, 1);                                                  // count=1
    dset(g0w, 32, 32, (unsigned)(unsigned long long)(uintptr_t)ldsDst);  // lds_addr
    dset(g0w, 64, 57, (unsigned long long)(uintptr_t)gsrc);              // global_addr
    dset(g0w, 126, 2, 2);                                                // type=image
    dset(g1w, 16, 2, 1);                                                 // data_size=2B
    dset(g1w, 48, 32, elems);                                            // tensor_dim0
    dset(g1w, 80, 32, rows);                                             // tensor_dim1
    dset(g1w, 112, 16, elems);                                           // tile_dim0
    dset(g1w, 128, 16, rows);                                            // tile_dim1
    dset(g1w, 160, 48, elems);                                           // dim0 stride
    u32x4 g0; i32x8 g1; i32x4 g2, g3;
#pragma unroll
    for (int i = 0; i < 4; ++i) { g0[i] = g0w[i]; g2[i] = (int)g2w[i]; g3[i] = (int)g3w[i]; }
#pragma unroll
    for (int i = 0; i < 8; ++i) g1[i] = (int)g1w[i];
#if defined(__clang_major__) && (__clang_major__ >= 23)
    i32x8 g4;
#pragma unroll
    for (int i = 0; i < 8; ++i) g4[i] = 0;
    __builtin_amdgcn_tensor_load_to_lds(g0, g1, g2, g3, g4, 0);
#else
    __builtin_amdgcn_tensor_load_to_lds(g0, g1, g2, g3, 0);
#endif
#else
    (void)gsrc; (void)ldsDst; (void)elems; (void)rows;
#endif
}

// Fallback linear staging (b128 global loads -> b128 ds stores)
__device__ inline void stage_copy(const _Float16* gsrc, _Float16* dst,
                                  int elems, int tid) {
    const v8h* s = (const v8h*)gsrc;
    v8h*       d = (v8h*)dst;
    for (int j = tid; j < (elems >> 3); j += 256) d[j] = s[j];
}

// One block = 256 threads = 8 waves; wave w computes 16 output rows.
// Weights pre-swizzled fragment-major in global memory. Small shapes preload
// all 27 offsets into LDS once; big shapes double-buffer per-offset slabs via
// TDM so the DMA for offset k+1 overlaps gather+WMMA of offset k.
template <int CINA, int CINB, int COUT>
__global__ __launch_bounds__(256) void spconv_wmma_kernel(ConvArgs a) {
    constexpr int CTOT = CINA + CINB;
    constexpr int KP   = (CTOT + 31) & ~31;
    constexpr int NKT  = KP / 32;
    constexpr int NNT  = COUT / 16;
    constexpr int WK   = KP * COUT;                 // halfs per offset slab
    constexpr bool PRE = (27 * WK * 2) <= 131072;   // preload all offsets?
    constexpr int NBUF = PRE ? 27 : 2;              // ping-pong when streaming

    __shared__ __attribute__((aligned(32))) _Float16 Wlds[NBUF * WK];
    __shared__ __attribute__((aligned(32))) _Float16 Alds[8][16 * KP];

    const int tid  = threadIdx.x;
    const int wave = tid >> 5;
    const int lane = tid & 31;
    const int r16  = lane & 15;
    const int hh   = lane >> 4;
    const int rowBase = (blockIdx.x * 8 + wave) * 16;

    _Float16* Aw = &Alds[wave][0];

    v8f acc[NNT];
    v8f zero = {};
#pragma unroll
    for (int i = 0; i < NNT; ++i) acc[i] = zero;

    if constexpr (PRE) {
#if HAVE_TDM
        if (wave == 0) {
            tdm_load_f16(a.W, &Wlds[0], WK, 27);
            __builtin_amdgcn_s_wait_tensorcnt(0);
        }
#else
        stage_copy(a.W, &Wlds[0], 27 * WK, tid);
#endif
        __syncthreads();
    } else {
#if HAVE_TDM
        if (wave == 0) tdm_load_f16(a.W, &Wlds[0], WK, 1);  // slab k=0 in flight
#endif
    }

    for (int k = 0; k < 27; ++k) {
        if constexpr (!PRE) {
#if HAVE_TDM
            // Wait for slab k (issued previous iteration), publish it, then
            // immediately launch slab k+1 into the other buffer. Its target
            // buffer was last read at iteration k-1, which the barrier retires.
            if (wave == 0) __builtin_amdgcn_s_wait_tensorcnt(0);
            __syncthreads();
            if (wave == 0 && k + 1 < 27)
                tdm_load_f16(a.W + (size_t)(k + 1) * WK, &Wlds[((k + 1) & 1) * WK],
                             WK, 1);
#else
            __syncthreads();  // previous readers done before overwrite
            stage_copy(a.W + (size_t)(k & 1) * 0 + (size_t)k * WK, &Wlds[0], WK, tid);
            if (k + 1 < 27)
                __builtin_prefetch(a.W + (size_t)(k + 1) * WK, 0, 0);
            __syncthreads();
#endif
        }

        // ---- gather A tile: 2 lanes per row, KP/2 consecutive channels each ----
        {
            constexpr int CH = KP / 2;
            const int r    = lane >> 1;
            const int row  = rowBase + r;
            const int cseg = (lane & 1) * CH;
            _Float16* dst  = &Aw[r * KP + cseg];
            int idx = -1;
            if (row < a.Nout) idx = a.nbr[(size_t)row * 27 + k];
            if (idx >= 0) {
#pragma unroll
                for (int c0 = 0; c0 < CH; ++c0) {
                    int c = cseg + c0;
                    float v = 0.f;
                    if (c < CTOT)
                        v = (c < CINA) ? a.featA[(size_t)idx * CINA + c]
                                       : a.featB[(size_t)idx * CINB + (c - CINA)];
                    dst[c0] = (_Float16)v;
                }
            } else {
#pragma unroll
                for (int c0 = 0; c0 < CH; ++c0) dst[c0] = (_Float16)0.f;
            }
        }
        // A slab is wave-private; LDS is in-order per wave. Only stop the
        // compiler from reordering, no cross-wave barrier needed here.
        __builtin_amdgcn_wave_barrier();

        // ---- WMMA accumulation ----
        const _Float16* wsl;
        if constexpr (PRE)
            wsl = &Wlds[k * WK];
        else
#if HAVE_TDM
            wsl = &Wlds[(k & 1) * WK];
#else
            wsl = &Wlds[0];
#endif
#pragma unroll
        for (int ks = 0; ks < NKT; ++ks) {
            union { v16h v; v8h h[2]; } af;
            const _Float16* ab = &Aw[r16 * KP + ks * 32 + (hh << 3)];
            af.h[0] = *(const v8h*)(ab);        // K = 8h .. 8h+7
            af.h[1] = *(const v8h*)(ab + 16);   // K = 16+8h .. 16+8h+7
#pragma unroll
            for (int nt = 0; nt < NNT; ++nt) {
                v16h bv = *(const v16h*)&wsl[((ks * NNT + nt) * 32 + lane) * 16];
                acc[nt] = __builtin_amdgcn_wmma_f32_16x16x32_f16(
                    false, af.v, false, bv, (short)0, acc[nt], false, false);
            }
        }
    }

    // ---- epilogue: BN scale/bias (+residual) + ReLU (+reduce(cat)) ----
#pragma unroll
    for (int nt = 0; nt < NNT; ++nt) {
#pragma unroll
        for (int i = 0; i < 8; ++i) {
            int row = rowBase + i + (hh << 3);   // C/D: M = i + 8*laneHalf
            int col = (nt << 4) + r16;
            if (row < a.Nout) {
                float v = acc[nt][i] * a.scale[col] + a.bias[col];
                if (a.mode == 1) v += a.res[(size_t)row * COUT + col];
                v = v > 0.f ? v : 0.f;
                if (a.mode == 2) {  // + reduce(cat): cat[n,2c] + cat[n,2c+1]
                    int j0 = 2 * col, j1 = j0 + 1;
                    float r0 = (j0 < CINA) ? a.featA[(size_t)row * CINA + j0]
                                           : a.featB[(size_t)row * CINB + (j0 - CINA)];
                    float r1 = (j1 < CINA) ? a.featA[(size_t)row * CINA + j1]
                                           : a.featB[(size_t)row * CINB + (j1 - CINA)];
                    v += r0 + r1;
                }
                a.out[(size_t)row * COUT + col] = v;
            }
        }
    }
}

template <int CINA, int CINB, int COUT>
static inline void conv(const float* fA, const float* fB, const int* nbr, int Nout,
                        const _Float16* W, const float* s, const float* b,
                        const float* res, int mode, float* out, hipStream_t st) {
    ConvArgs a;
    a.featA = fA; a.featB = fB; a.nbr = nbr; a.W = W;
    a.scale = s; a.bias = b; a.res = res; a.out = out;
    a.Nout = Nout; a.mode = mode;
    dim3 grid((unsigned)((Nout + 127) / 128));
    spconv_wmma_kernel<CINA, CINB, COUT><<<grid, 256, 0, st>>>(a);
}

extern "C" void kernel_launch(void* const* d_in, const int* in_sizes, int n_in,
                              void* d_out, int out_size, void* d_ws, size_t ws_size,
                              hipStream_t stream) {
    const float* vox    = (const float*)d_in[0];   // N1 x 4
    const float* Win    = (const float*)d_in[1];   // 27x4x32
    const float* W32    = (const float*)d_in[2];   // 10x27x32x32
    const float* W64    = (const float*)d_in[3];   // 10x27x64x64
    const float* Wd3    = (const float*)d_in[4];   // 27x32x64
    const float* W6432  = (const float*)d_in[5];   // 3x27x64x32
    const float* W12864 = (const float*)d_in[6];   // 2x27x128x64
    const float* bn32   = (const float*)d_in[7];   // 14x2x32
    const float* bn64   = (const float*)d_in[8];   // 13x2x64
    const int* nbr1  = (const int*)d_in[9];
    const int* nbr2  = (const int*)d_in[10];
    const int* nbr3  = (const int*)d_in[11];
    const int* nbr4  = (const int*)d_in[12];
    const int* nbrd2 = (const int*)d_in[13];
    const int* nbrd3 = (const int*)d_in[14];
    const int* nbrd4 = (const int*)d_in[15];
    const int* nbri4 = (const int*)d_in[16];
    const int* nbri3 = (const int*)d_in[17];
    const int* nbri2 = (const int*)d_in[18];

    const int N1 = in_sizes[9]  / 27;
    const int N2 = in_sizes[10] / 27;
    const int N3 = in_sizes[11] / 27;
    const int N4 = in_sizes[12] / 27;

    // padded, fragment-major f16 slab sizes (27 offsets each)
    const int nWin    = 27 * 32 * 32;          // KP=32 (Cin 4 padded)
    const int nW32    = 10 * 27 * 32 * 32;
    const int nW64    = 10 * 27 * 64 * 64;
    const int nWd3    = 27 * 32 * 64;
    const int nW6432  = 3 * 27 * 64 * 32;
    const int nW12864 = 2 * 27 * 128 * 64;

    char* ws = (char*)d_ws;
    size_t off = 0;
    auto allocH = [&](size_t n) { _Float16* p = (_Float16*)(ws + off);
                                  off = (off + n * 2 + 31) & ~(size_t)31; return p; };
    auto allocF = [&](size_t n) { float* p = (float*)(ws + off);
                                  off = (off + n * 4 + 31) & ~(size_t)31; return p; };

    _Float16* hWin    = allocH(nWin);
    _Float16* hW32    = allocH(nW32);
    _Float16* hW64    = allocH(nW64);
    _Float16* hWd3    = allocH(nWd3);
    _Float16* hW6432  = allocH(nW6432);
    _Float16* hW12864 = allocH(nW12864);

    float* x0  = allocF((size_t)N1 * 32);
    float* x1  = allocF((size_t)N1 * 32);
    float* t1h = allocF((size_t)N1 * 32);
    float* t1  = allocF((size_t)N1 * 32);
    float* h2a = allocF((size_t)N2 * 32);
    float* h2b = allocF((size_t)N2 * 32);
    float* x2  = allocF((size_t)N2 * 32);
    float* t2  = allocF((size_t)N2 * 32);
    float* h3a = allocF((size_t)N3 * 64);
    float* h3b = allocF((size_t)N3 * 64);
    float* x3  = allocF((size_t)N3 * 64);
    float* t3  = allocF((size_t)N3 * 64);
    float* h4a = allocF((size_t)N4 * 64);
    float* h4b = allocF((size_t)N4 * 64);
    float* x4  = allocF((size_t)N4 * 64);
    float* dec4= allocF((size_t)N4 * 64);
    (void)ws_size; (void)n_in; (void)out_size;

    // ---- repack weights: f32 row-major -> f16 fragment-major (swizzled) ----
    auto rp_grid = [](int n) { return dim3((unsigned)((n + 255) / 256)); };
    repack_kernel<4,  32><<<rp_grid(nWin),    256, 0, stream>>>(Win,    hWin,    27);
    repack_kernel<32, 32><<<rp_grid(nW32),    256, 0, stream>>>(W32,    hW32,    10 * 27);
    repack_kernel<64, 64><<<rp_grid(nW64),    256, 0, stream>>>(W64,    hW64,    10 * 27);
    repack_kernel<32, 64><<<rp_grid(nWd3),    256, 0, stream>>>(Wd3,    hWd3,    27);
    repack_kernel<64, 32><<<rp_grid(nW6432),  256, 0, stream>>>(W6432,  hW6432,  3 * 27);
    repack_kernel<128,64><<<rp_grid(nW12864), 256, 0, stream>>>(W12864, hW12864, 2 * 27);

    auto s32 = [&](int i) { return bn32 + (size_t)(2 * i + 0) * 32; };
    auto b32 = [&](int i) { return bn32 + (size_t)(2 * i + 1) * 32; };
    auto s64 = [&](int i) { return bn64 + (size_t)(2 * i + 0) * 64; };
    auto b64 = [&](int i) { return bn64 + (size_t)(2 * i + 1) * 64; };
    auto w32 = [&](int i) { return hW32 + (size_t)i * 27 * 32 * 32; };
    auto w64 = [&](int i) { return hW64 + (size_t)i * 27 * 64 * 64; };
    auto w6432  = [&](int i) { return hW6432 + (size_t)i * 27 * 64 * 32; };
    auto w12864 = [&](int i) { return hW12864 + (size_t)i * 27 * 128 * 64; };

    // ================= encoder =================
    conv<4, 0, 32>(vox, nullptr, nbr1,  N1, hWin,   s32(0), b32(0), nullptr, 0, x0,  stream);
    conv<32,0, 32>(x0,  nullptr, nbr1,  N1, w32(0), s32(1), b32(1), nullptr, 0, x1,  stream);
    conv<32,0, 32>(x1,  nullptr, nbrd2, N2, w32(1), s32(2), b32(2), nullptr, 0, h2a, stream);
    conv<32,0, 32>(h2a, nullptr, nbr2,  N2, w32(2), s32(3), b32(3), nullptr, 0, h2b, stream);
    conv<32,0, 32>(h2b, nullptr, nbr2,  N2, w32(3), s32(4), b32(4), nullptr, 0, x2,  stream);
    conv<32,0, 64>(x2,  nullptr, nbrd3, N3, hWd3,   s64(0), b64(0), nullptr, 0, h3a, stream);
    conv<64,0, 64>(h3a, nullptr, nbr3,  N3, w64(0), s64(1), b64(1), nullptr, 0, h3b, stream);
    conv<64,0, 64>(h3b, nullptr, nbr3,  N3, w64(1), s64(2), b64(2), nullptr, 0, x3,  stream);
    conv<64,0, 64>(x3,  nullptr, nbrd4, N4, w64(2), s64(3), b64(3), nullptr, 0, h4a, stream);
    conv<64,0, 64>(h4a, nullptr, nbr4,  N4, w64(3), s64(4), b64(4), nullptr, 0, h4b, stream);
    conv<64,0, 64>(h4b, nullptr, nbr4,  N4, w64(4), s64(5), b64(5), nullptr, 0, x4,  stream);

    // ================= level-4 decode =================
    float* t4h = h4a; float* t4 = h4b;
    conv<64,0, 64>(x4,  nullptr, nbr4, N4, w64(5), s64(6), b64(6), nullptr, 0, t4h, stream);
    conv<64,0, 64>(t4h, nullptr, nbr4, N4, w64(6), s64(7), b64(7), x4,      1, t4,  stream);
    conv<64,64,64>(x4,  t4,      nbr4, N4, w12864(0), s64(11), b64(11), nullptr, 2, dec4, stream);
    float* xu4 = h3a;
    conv<64,0, 64>(dec4, nullptr, nbri4, N3, w64(7), s64(8), b64(8), nullptr, 0, xu4, stream);

    // ================= level-3 decode =================
    float* t3h = h3b;
    conv<64,0, 64>(x3,  nullptr, nbr3, N3, w64(8), s64(9),  b64(9),  nullptr, 0, t3h, stream);
    conv<64,0, 64>(t3h, nullptr, nbr3, N3, w64(9), s64(10), b64(10), x3,      1, t3,  stream);
    float* dec3 = h3b;
    conv<64,64,64>(xu4, t3,      nbr3, N3, w12864(1), s64(12), b64(12), nullptr, 2, dec3, stream);
    float* xu3 = h2a;
    conv<64,0, 32>(dec3, nullptr, nbri3, N2, w6432(0), s32(11), b32(11), nullptr, 0, xu3, stream);

    // ================= level-2 decode =================
    float* t2h = h2b;
    conv<32,0, 32>(x2,  nullptr, nbr2, N2, w32(4), s32(5), b32(5), nullptr, 0, t2h, stream);
    conv<32,0, 32>(t2h, nullptr, nbr2, N2, w32(5), s32(6), b32(6), x2,      1, t2,  stream);
    float* dec2 = h2b;
    conv<32,32,32>(xu3, t2,      nbr2, N2, w6432(1), s32(12), b32(12), nullptr, 2, dec2, stream);
    float* xu2 = x0;
    conv<32,0, 32>(dec2, nullptr, nbri2, N1, w32(6), s32(7), b32(7), nullptr, 0, xu2, stream);

    // ================= level-1 decode + head =================
    conv<32,0, 32>(x1,  nullptr, nbr1, N1, w32(7), s32(8), b32(8), nullptr, 0, t1h, stream);
    conv<32,0, 32>(t1h, nullptr, nbr1, N1, w32(8), s32(9), b32(9), x1,      1, t1,  stream);
    float* dec1 = t1h;
    conv<32,32,32>(xu2, t1,      nbr1, N1, w6432(2), s32(13), b32(13), nullptr, 2, dec1, stream);
    conv<32,0, 32>(dec1, nullptr, nbr1, N1, w32(9), s32(10), b32(10), nullptr, 0, (float*)d_out, stream);
}
